// GumbelSoftmaxModule4_3272765079609
// MI455X (gfx1250) — matrix operand order repaired and checked by reading, not compiled, not run on hardware
//
#include <hip/hip_runtime.h>
#include <hip/hip_bf16.h>

// ---- problem constants (from reference) ----
#define KCODES 512
#define DFULL  512
#define NGRP   4
#define SUBD   128
#define BB     16
#define TT     2048
#define NTOK   (BB*TT)          // 32768
#define QUANT_ELEMS ((size_t)BB*DFULL*TT)  // 16777216

typedef __attribute__((ext_vector_type(16))) _Float16 v16h;
typedef __attribute__((ext_vector_type(8)))  _Float16 v8h;
typedef __attribute__((ext_vector_type(4)))  _Float16 v4h;
typedef __attribute__((ext_vector_type(8)))  float    v8f;

static __device__ __forceinline__ v16h combine16(v8h lo, v8h hi) {
    v16h r;
#pragma unroll
    for (int i = 0; i < 8; ++i) { r[i] = lo[i]; r[i + 8] = hi[i]; }
    return r;
}

static __device__ __forceinline__ v8f wmma_f16(v16h a, v16h b, v8f c) {
    // D(f32 16x16) = A(f16 16x32) * B(f16 32x16) + C
    return __builtin_amdgcn_wmma_f32_16x16x32_f16(false, a, false, b, (short)0, c, false, false);
}

// ---------------------------------------------------------------------------
// Setup: W (f32 [K][SUBD]) -> Wh f16 [g][k][d], WhT f16 [g][d][k], csq[g][k]
// grid: (512, 4) blocks of 128 threads; blockIdx.x = k, blockIdx.y = g
// ---------------------------------------------------------------------------
__global__ void vq_prep(const float* __restrict__ W1, const float* __restrict__ W2,
                        const float* __restrict__ W3, const float* __restrict__ W4,
                        _Float16* __restrict__ Wh, _Float16* __restrict__ WhT,
                        float* __restrict__ csq) {
    int k = blockIdx.x, g = blockIdx.y;
    int d = threadIdx.x;             // 0..127
    const float* Wg = (g == 0) ? W1 : (g == 1) ? W2 : (g == 2) ? W3 : W4;
    float v = Wg[k * SUBD + d];
    Wh [((size_t)g * KCODES + k) * SUBD + d] = (_Float16)v;
    WhT[((size_t)g * SUBD  + d) * KCODES + k] = (_Float16)v;
    // block reduce sum of squares (4 waves of 32)
    float s = v * v;
#pragma unroll
    for (int off = 16; off; off >>= 1) s += __shfl_down(s, off, 32);
    __shared__ float r4[4];
    int lane = threadIdx.x & 31, wv = threadIdx.x >> 5;
    if (lane == 0) r4[wv] = s;
    __syncthreads();
    if (threadIdx.x == 0) csq[g * KCODES + k] = r4[0] + r4[1] + r4[2] + r4[3];
}

__global__ void vq_zero(float* __restrict__ avg) {
    int i = blockIdx.x * blockDim.x + threadIdx.x;
    if (i < NGRP * KCODES) avg[i] = 0.0f;
}

// ---------------------------------------------------------------------------
// Fused main kernel: distances (WMMA) + gumbel softmax + quant (WMMA)
// grid: (512, 4), block 128 threads (4 wave32).  blockIdx.x = token tile of 64,
// blockIdx.y = group.  Dynamic LDS: 131072 (cross/enc) + 16384 (x f16) + 2048
// ---------------------------------------------------------------------------
__global__ void vq_main(const float* __restrict__ x,
                        const float* __restrict__ gum,
                        const _Float16* __restrict__ Wh,
                        const _Float16* __restrict__ WhT,
                        const float* __restrict__ csq,
                        float* __restrict__ avg,
                        float* __restrict__ out) {
    extern __shared__ __align__(16) char smem[];
    float*    lgtBase = (float*)smem;                      // [4][16][512] f32
    _Float16* xs      = (_Float16*)(smem + 131072);        // [64 tok][128 d] f16
    float*    avgLds  = (float*)(smem + 147456);           // [512]

    const int g   = blockIdx.y;
    const int n0  = blockIdx.x * 64;          // 64 tokens per block, within one b
    const int b   = n0 >> 11;                 // n0 / 2048
    const int t0  = n0 & 2047;
    const int tid = threadIdx.x;
    const int wv  = tid >> 5;
    const int lane = tid & 31;
    const int lrow = lane & 15;               // M / N column index
    const int lhi  = lane >> 4;               // 0 or 1

    // zero per-block avg accumulator
    for (int i = tid; i < KCODES; i += 128) avgLds[i] = 0.0f;

    // stage x tile: float4 along t (global_load_b128), store token-major f16
    {
        const float* xb = x + ((size_t)b * DFULL + g * SUBD) * TT + t0;
#pragma unroll 4
        for (int it = 0; it < 16; ++it) {
            int idx = it * 128 + tid;         // idx = d*16 + t4
            int d = idx >> 4, t4 = idx & 15;
            float4 v = *(const float4*)(xb + (size_t)d * TT + t4 * 4);
            xs[(t4 * 4 + 0) * SUBD + d] = (_Float16)v.x;
            xs[(t4 * 4 + 1) * SUBD + d] = (_Float16)v.y;
            xs[(t4 * 4 + 2) * SUBD + d] = (_Float16)v.z;
            xs[(t4 * 4 + 3) * SUBD + d] = (_Float16)v.w;
        }
    }
    __syncthreads();

    // prefetch this wave's gumbel strip (16 tok x 512 k x 4B = 32KB = 256 lines)
    const size_t gumBase = ((size_t)g * NTOK + n0 + wv * 16) * KCODES;
    {
        const char* gp = (const char*)(gum + gumBase);
#pragma unroll
        for (int j = 0; j < 8; ++j)
            __builtin_prefetch(gp + (size_t)(lane + 32 * j) * 128, 0, 3);
    }

    // ---- Phase 1: cross strip [16 tok][512 k] per wave (pure WMMA loop) ----
    float* lgt = lgtBase + wv * (16 * KCODES);
    v16h afrag[4];
#pragma unroll
    for (int dc = 0; dc < 4; ++dc) {
        int tl = wv * 16 + lrow;
        int dstart = dc * 32 + lhi * 16;
        v8h lo = *(const v8h*)&xs[tl * SUBD + dstart];
        v8h hi = *(const v8h*)&xs[tl * SUBD + dstart + 8];
        afrag[dc] = combine16(lo, hi);
    }
    for (int kt = 0; kt < 32; ++kt) {
        v8f acc = {0.f, 0.f, 0.f, 0.f, 0.f, 0.f, 0.f, 0.f};
        int kcol = kt * 16 + lrow;
        const _Float16* wrow = Wh + ((size_t)g * KCODES + kcol) * SUBD + lhi * 16;
#pragma unroll
        for (int dc = 0; dc < 4; ++dc) {
            v16h bfrag = *(const v16h*)(wrow + dc * 32);   // contiguous 32B per lane
            acc = wmma_f16(afrag[dc], bfrag, acc);
        }
#pragma unroll
        for (int r = 0; r < 8; ++r)
            lgt[(r + lhi * 8) * KCODES + kcol] = acc[r];   // raw cross term
    }

    // ---- Phase 2: logits + softmax, fully coalesced gumbel/csq (b128) ----
    // lane owns k = j*128 + lane*4 + {0..3}
    _Float16* encw = (_Float16*)(smem + wv * 32768);       // [16][512] f16
    float4 csq4[4];
#pragma unroll
    for (int j = 0; j < 4; ++j)
        csq4[j] = *(const float4*)(csq + g * KCODES + j * 128 + lane * 4);
    float4 kpart[4];
#pragma unroll
    for (int j = 0; j < 4; ++j) kpart[j] = make_float4(0.f, 0.f, 0.f, 0.f);

    for (int m = 0; m < 16; ++m) {
        const float*  row = lgt + m * KCODES;
        const float*  gr  = gum + gumBase + (size_t)m * KCODES;
        float4 vals[4];
        float mx = -3.4e38f;
#pragma unroll
        for (int j = 0; j < 4; ++j) {
            int k = j * 128 + lane * 4;
            float4 cr = *(const float4*)(row + k);
            float4 gv = *(const float4*)(gr + k);          // global_load_b128, coalesced
            vals[j].x = cr.x + 0.5f * (gv.x - csq4[j].x);
            vals[j].y = cr.y + 0.5f * (gv.y - csq4[j].y);
            vals[j].z = cr.z + 0.5f * (gv.z - csq4[j].z);
            vals[j].w = cr.w + 0.5f * (gv.w - csq4[j].w);
            mx = fmaxf(mx, fmaxf(fmaxf(vals[j].x, vals[j].y), fmaxf(vals[j].z, vals[j].w)));
        }
#pragma unroll
        for (int off = 16; off; off >>= 1) mx = fmaxf(mx, __shfl_xor(mx, off, 32));
        float s = 0.0f;
#pragma unroll
        for (int j = 0; j < 4; ++j) {
            vals[j].x = __expf(vals[j].x - mx); s += vals[j].x;
            vals[j].y = __expf(vals[j].y - mx); s += vals[j].y;
            vals[j].z = __expf(vals[j].z - mx); s += vals[j].z;
            vals[j].w = __expf(vals[j].w - mx); s += vals[j].w;
        }
#pragma unroll
        for (int off = 16; off; off >>= 1) s += __shfl_xor(s, off, 32);
        float inv = __frcp_rn(s);
        _Float16* erow = encw + m * KCODES;
#pragma unroll
        for (int j = 0; j < 4; ++j) {
            float4 p = make_float4(vals[j].x * inv, vals[j].y * inv,
                                   vals[j].z * inv, vals[j].w * inv);
            v4h ph = {(_Float16)p.x, (_Float16)p.y, (_Float16)p.z, (_Float16)p.w};
            *(v4h*)&erow[j * 128 + lane * 4] = ph;          // ds_store_b64
            kpart[j].x += p.x; kpart[j].y += p.y; kpart[j].z += p.z; kpart[j].w += p.w;
        }
    }
#pragma unroll
    for (int j = 0; j < 4; ++j) {
        int k = j * 128 + lane * 4;
        atomicAdd(&avgLds[k + 0], kpart[j].x);
        atomicAdd(&avgLds[k + 1], kpart[j].y);
        atomicAdd(&avgLds[k + 2], kpart[j].z);
        atomicAdd(&avgLds[k + 3], kpart[j].w);
    }
    __syncthreads();
    for (int i = tid; i < KCODES; i += 128) atomicAdd(&avg[g * KCODES + i], avgLds[i]);

    // ---- Phase 3: quant = enc(f16) @ W(f16), 16 tok x 128 d per wave ----
    v8f qacc[8];
#pragma unroll
    for (int dt = 0; dt < 8; ++dt) qacc[dt] = (v8f){0.f,0.f,0.f,0.f,0.f,0.f,0.f,0.f};
    for (int kc = 0; kc < 16; ++kc) {
        int kstart = kc * 32 + lhi * 16;
        v8h lo = *(const v8h*)&encw[lrow * KCODES + kstart];
        v8h hi = *(const v8h*)&encw[lrow * KCODES + kstart + 8];
        v16h a = combine16(lo, hi);
#pragma unroll
        for (int dt = 0; dt < 8; ++dt) {
            const _Float16* wcol = WhT + ((size_t)g * SUBD + dt * 16 + lrow) * KCODES
                                       + kc * 32 + lhi * 16;
            v16h bfrag = *(const v16h*)wcol;               // contiguous 32B per lane
            qacc[dt] = wmma_f16(a, bfrag, qacc[dt]);
        }
    }
    // stage result for coalesced writeout
    float* qstage = (float*)(smem + wv * 32768 + 16384);   // [16][128]
#pragma unroll
    for (int dt = 0; dt < 8; ++dt)
#pragma unroll
        for (int r = 0; r < 8; ++r)
            qstage[(r + lhi * 8) * SUBD + dt * 16 + lrow] = qacc[dt][r];
    __syncthreads();

    // writeout: out[b][g*128+dd][t0 + 4*t4 + c] as float4 (global_store_b128)
    {
        float* ob = out + ((size_t)b * DFULL + g * SUBD) * TT + t0;
#pragma unroll 4
        for (int it = 0; it < 16; ++it) {
            int idx = it * 128 + tid;          // idx = dd*16 + t4
            int dd = idx >> 4, t4 = idx & 15;
            float4 v;
            int tl = t4 * 4;
            v.x = ((const float*)(smem + ((tl + 0) >> 4) * 32768 + 16384))[((tl + 0) & 15) * SUBD + dd];
            v.y = ((const float*)(smem + ((tl + 1) >> 4) * 32768 + 16384))[((tl + 1) & 15) * SUBD + dd];
            v.z = ((const float*)(smem + ((tl + 2) >> 4) * 32768 + 16384))[((tl + 2) & 15) * SUBD + dd];
            v.w = ((const float*)(smem + ((tl + 3) >> 4) * 32768 + 16384))[((tl + 3) & 15) * SUBD + dd];
            *(float4*)(ob + (size_t)dd * TT + tl) = v;
        }
    }
}

// ---------------------------------------------------------------------------
// Perplexity: perp = sum_g exp(-sum_k p*log(p+1e-10)), p = avg/N
// ---------------------------------------------------------------------------
__global__ void vq_perp(const float* __restrict__ avg, float* __restrict__ out) {
    __shared__ float red[256];
    int tid = threadIdx.x;
    float perp = 0.0f;
    for (int g = 0; g < NGRP; ++g) {
        float part = 0.0f;
        for (int k = tid; k < KCODES; k += 256) {
            float p = avg[g * KCODES + k] * (1.0f / (float)NTOK);
            part += p * __logf(p + 1e-10f);
        }
        red[tid] = part;
        __syncthreads();
        for (int off = 128; off; off >>= 1) {
            if (tid < off) red[tid] += red[tid + off];
            __syncthreads();
        }
        if (tid == 0) perp += __expf(-red[0]);
        __syncthreads();
    }
    if (tid == 0) out[QUANT_ELEMS] = perp;
}

// ---------------------------------------------------------------------------
extern "C" void kernel_launch(void* const* d_in, const int* in_sizes, int n_in,
                              void* d_out, int out_size, void* d_ws, size_t ws_size,
                              hipStream_t stream) {
    const float* x   = (const float*)d_in[0];
    const float* W1  = (const float*)d_in[1];
    const float* W2  = (const float*)d_in[2];
    const float* W3  = (const float*)d_in[3];
    const float* W4  = (const float*)d_in[4];
    const float* gum = (const float*)d_in[5];
    float* out = (float*)d_out;

    char* ws = (char*)d_ws;
    _Float16* Wh  = (_Float16*)(ws);                 // 4*512*128*2 = 524288 B
    _Float16* WhT = (_Float16*)(ws + 524288);        // 524288 B
    float*    csq = (float*)(ws + 1048576);          // 8192 B
    float*    avg = (float*)(ws + 1056768);          // 8192 B

    vq_zero<<<8, 256, 0, stream>>>(avg);
    vq_prep<<<dim3(KCODES, NGRP), 128, 0, stream>>>(W1, W2, W3, W4, Wh, WhT, csq);
    const size_t lds = 131072 + 16384 + 2048;        // 149504 B
    vq_main<<<dim3(NTOK / 64, NGRP), 128, lds, stream>>>(x, gum, Wh, WhT, csq, avg, out);
    vq_perp<<<1, 256, 0, stream>>>(avg, out);
}
